// VectorQuantizerEMA_56736517980778
// MI455X (gfx1250) — matrix operand order repaired and checked by reading, not compiled, not run on hardware
//
#include <hip/hip_runtime.h>
#include <hip/hip_bf16.h>
#include <stdint.h>

#define NUM_EMB 8192
#define DIM     512
#define NROWS   32768
#define MTILE   64            // x rows per workgroup (4 waves x 16)
#define KBCODES 32            // codes per staged e-tile (wave N-tile = 32)
#define NKB     (NUM_EMB / KBCODES)
#define LROW    520           // padded LDS row stride in elements (1040 B)

typedef __bf16 bf16_t;
typedef __attribute__((ext_vector_type(16))) __bf16 v16bf;
typedef __attribute__((ext_vector_type(8)))  __bf16 v8bf;
typedef __attribute__((ext_vector_type(4)))  __bf16 v4bf;
typedef __attribute__((ext_vector_type(8)))  float  v8f;
typedef unsigned int u32x4 __attribute__((ext_vector_type(4)));
typedef int          i32x8 __attribute__((ext_vector_type(8)));
typedef int          i32x4 __attribute__((ext_vector_type(4)));

// ---------------------------------------------------------------------------
// TDM: DMA one [KBCODES x 512] bf16 plane from global into LDS, inserting
// 16 B of padding every 1024 B row (pad_interval=7 -> 256 dwords,
// pad_amount=3 -> 4 dwords) so LDS rows land at LROW*2 = 1040 B stride.
// ---------------------------------------------------------------------------
__device__ __forceinline__ void tdm_issue(unsigned lds_off, const bf16_t* gsrc) {
  unsigned long long ga = (unsigned long long)(uintptr_t)gsrc;
  u32x4 g0;
  g0[0] = 1u;                                            // count=1, user mode
  g0[1] = lds_off;                                       // lds_addr
  g0[2] = (unsigned)ga;                                  // global_addr[31:0]
  g0[3] = (unsigned)((ga >> 32) & 0x1FFFFFFu) | (2u << 30); // addr[56:32] | type=2
  i32x8 g1;
  g1[0] = (int)((1u << 16) | (1u << 20) | (7u << 22) | (3u << 25)); // 2B elems, pad en, 256dw/4dw
  g1[1] = (int)(512u << 16);   // tensor_dim0 = 512 (bits 79:48)
  g1[2] = (int)(8192u << 16);  // tensor_dim1 = 8192 (bits 111:80, low16 here)
  g1[3] = (int)(512u << 16);   // tensor_dim1 hi = 0 | tile_dim0 = 512
  g1[4] = (int)KBCODES;        // tile_dim1 = 32, tile_dim2 = 0
  g1[5] = 512;                 // tensor_dim0_stride low 32 = 512 elements
  g1[6] = 0;                   // stride hi / dim1 stride
  g1[7] = 0;
  i32x4 z4 = {0, 0, 0, 0};
#if defined(__clang_major__) && (__clang_major__ >= 23)
  i32x8 z8 = {0, 0, 0, 0, 0, 0, 0, 0};
  __builtin_amdgcn_tensor_load_to_lds(g0, g1, z4, z4, z8, 0);
#else
  __builtin_amdgcn_tensor_load_to_lds(g0, g1, z4, z4, 0);
#endif
}

// ---------------------------------------------------------------------------
// Prep: E fp32 -> bf16 hi/lo planes + ||e||^2; zero counts & loss accumulator.
// grid = NUM_EMB blocks x 128 threads (4 dims/thread).
// ---------------------------------------------------------------------------
__global__ __launch_bounds__(128) void vq_prep(const float* __restrict__ e,
                                               bf16_t* __restrict__ ehi,
                                               bf16_t* __restrict__ elo,
                                               float* __restrict__ esq,
                                               int* __restrict__ counts,
                                               float* __restrict__ loss) {
  const int k = blockIdx.x;
  const int t = threadIdx.x;
  float4 v = ((const float4*)(e + (size_t)k * DIM))[t];
  float xs[4] = {v.x, v.y, v.z, v.w};
  v4bf h, l;
  float sq = 0.f;
#pragma unroll
  for (int c = 0; c < 4; ++c) {
    float f = xs[c];
    bf16_t hb = (bf16_t)f;
    h[c] = hb;
    l[c] = (bf16_t)(f - (float)hb);
    sq = fmaf(f, f, sq);
  }
  ((v4bf*)(ehi + (size_t)k * DIM))[t] = h;
  ((v4bf*)(elo + (size_t)k * DIM))[t] = l;
#pragma unroll
  for (int o = 16; o >= 1; o >>= 1) sq += __shfl_down(sq, o, 32);
  __shared__ float r[4];
  if ((t & 31) == 0) r[t >> 5] = sq;
  __syncthreads();
  if (t == 0) {
    esq[k] = r[0] + r[1] + r[2] + r[3];
    counts[k] = 0;
    if (k == 0) *loss = 0.f;
  }
}

// ---------------------------------------------------------------------------
// Fused GEMM + argmin.  grid = NROWS/MTILE blocks x 128 threads (4 waves).
// Each wave owns a 16(M) x 32(N) register tile (two v8f accumulators) so the
// A fragments (x) are reused across both N-subtiles -> 12 ds_load_b128 per
// 6 WMMAs instead of 8 per 3.
// dist = ||e||^2 - 2 x.e ; x.e via bf16 hi/lo split (hi*hi + hi*lo + lo*hi).
// ---------------------------------------------------------------------------
__global__ __launch_bounds__(128) void vq_argmin(const float* __restrict__ x,
                                                 const bf16_t* __restrict__ ehi_g,
                                                 const bf16_t* __restrict__ elo_g,
                                                 const float* __restrict__ esq_g,
                                                 int* __restrict__ idx_g,
                                                 float* __restrict__ out_idx) {
  __shared__ __attribute__((aligned(16))) bf16_t s_xhi[MTILE * LROW];
  __shared__ __attribute__((aligned(16))) bf16_t s_xlo[MTILE * LROW];
  __shared__ __attribute__((aligned(16))) bf16_t s_ehi[2][KBCODES * LROW];
  __shared__ __attribute__((aligned(16))) bf16_t s_elo[2][KBCODES * LROW];

  const int tid  = threadIdx.x;
  const int wave = tid >> 5;
  const int lane = tid & 31;
  const int half = lane >> 4;
  const int lx   = lane & 15;
  const int m0   = blockIdx.x * MTILE;

  // Stage x tile (64 x 512 fp32) into LDS as bf16 hi/lo with padded rows.
  for (int i = tid; i < MTILE * (DIM / 4); i += 128) {
    int m  = i >> 7;            // 128 float4 per row
    int dq = (i & 127) << 2;
    float4 v = *(const float4*)(x + (size_t)(m0 + m) * DIM + dq);
    float xs[4] = {v.x, v.y, v.z, v.w};
    v4bf h, l;
#pragma unroll
    for (int c = 0; c < 4; ++c) {
      float f = xs[c];
      bf16_t hb = (bf16_t)f;
      h[c] = hb;
      l[c] = (bf16_t)(f - (float)hb);
    }
    *(v4bf*)(s_xhi + m * LROW + dq) = h;
    *(v4bf*)(s_xlo + m * LROW + dq) = l;
  }

  // Prologue: DMA first e-tile (hi+lo planes) into buffer 0.
  if (wave == 0) {
    tdm_issue((unsigned)(uintptr_t)(void*)&s_ehi[0][0], ehi_g);
    tdm_issue((unsigned)(uintptr_t)(void*)&s_elo[0][0], elo_g);
  }

  float best[8];
  int   bidx[8];
#pragma unroll
  for (int j = 0; j < 8; ++j) { best[j] = 3.4e38f; bidx[j] = 0; }

  // A-fragment base (16-bit A 16x32 layout): lane = M row, halves split K.
  const bf16_t* arow_h = s_xhi + (wave * 16 + lx) * LROW + 8 * half;
  const bf16_t* arow_l = s_xlo + (wave * 16 + lx) * LROW + 8 * half;

  for (int kb = 0; kb < NKB; ++kb) {
    const int buf = kb & 1;
    if (wave == 0) {
      if (kb + 1 < NKB) {
        const size_t roff = (size_t)(kb + 1) * KBCODES * DIM;
        tdm_issue((unsigned)(uintptr_t)(void*)&s_ehi[buf ^ 1][0], ehi_g + roff);
        tdm_issue((unsigned)(uintptr_t)(void*)&s_elo[buf ^ 1][0], elo_g + roff);
        __builtin_amdgcn_s_wait_tensorcnt(2);  // current tile's 2 DMAs done
      } else {
        __builtin_amdgcn_s_wait_tensorcnt(0);
      }
    }
    __syncthreads();  // e-tile `buf` visible to all waves

    v8f acc0 = (v8f)0.f;                 // codes kb*32 + [0,16)
    v8f acc1 = (v8f)0.f;                 // codes kb*32 + [16,32)
    // B-fragment bases (16-bit B 32x16 layout): lane = N column (= code row
    // in the staged tile), halves split K.
    const bf16_t* b0h = &s_ehi[buf][0] + lx * LROW + 8 * half;
    const bf16_t* b0l = &s_elo[buf][0] + lx * LROW + 8 * half;
    const bf16_t* b1h = b0h + 16 * LROW;
    const bf16_t* b1l = b0l + 16 * LROW;
#pragma unroll
    for (int ks = 0; ks < DIM / 32; ++ks) {
      const int d0 = ks * 32;
      v8bf ah0 = *(const v8bf*)(arow_h + d0);
      v8bf ah1 = *(const v8bf*)(arow_h + d0 + 16);
      v8bf al0 = *(const v8bf*)(arow_l + d0);
      v8bf al1 = *(const v8bf*)(arow_l + d0 + 16);
      v16bf Ah = __builtin_shufflevector(ah0, ah1, 0,1,2,3,4,5,6,7,8,9,10,11,12,13,14,15);
      v16bf Al = __builtin_shufflevector(al0, al1, 0,1,2,3,4,5,6,7,8,9,10,11,12,13,14,15);

      v8bf p0 = *(const v8bf*)(b0h + d0);
      v8bf p1 = *(const v8bf*)(b0h + d0 + 16);
      v8bf q0 = *(const v8bf*)(b0l + d0);
      v8bf q1 = *(const v8bf*)(b0l + d0 + 16);
      v16bf B0h = __builtin_shufflevector(p0, p1, 0,1,2,3,4,5,6,7,8,9,10,11,12,13,14,15);
      v16bf B0l = __builtin_shufflevector(q0, q1, 0,1,2,3,4,5,6,7,8,9,10,11,12,13,14,15);
      acc0 = __builtin_amdgcn_wmma_f32_16x16x32_bf16(false, Ah, false, B0h, (short)0, acc0, false, false);
      acc0 = __builtin_amdgcn_wmma_f32_16x16x32_bf16(false, Ah, false, B0l, (short)0, acc0, false, false);
      acc0 = __builtin_amdgcn_wmma_f32_16x16x32_bf16(false, Al, false, B0h, (short)0, acc0, false, false);

      v8bf r0 = *(const v8bf*)(b1h + d0);
      v8bf r1 = *(const v8bf*)(b1h + d0 + 16);
      v8bf s0 = *(const v8bf*)(b1l + d0);
      v8bf s1 = *(const v8bf*)(b1l + d0 + 16);
      v16bf B1h = __builtin_shufflevector(r0, r1, 0,1,2,3,4,5,6,7,8,9,10,11,12,13,14,15);
      v16bf B1l = __builtin_shufflevector(s0, s1, 0,1,2,3,4,5,6,7,8,9,10,11,12,13,14,15);
      acc1 = __builtin_amdgcn_wmma_f32_16x16x32_bf16(false, Ah, false, B1h, (short)0, acc1, false, false);
      acc1 = __builtin_amdgcn_wmma_f32_16x16x32_bf16(false, Ah, false, B1l, (short)0, acc1, false, false);
      acc1 = __builtin_amdgcn_wmma_f32_16x16x32_bf16(false, Al, false, B1h, (short)0, acc1, false, false);
    }
    __syncthreads();  // all waves done with e-tile `buf` before it is refilled

    const int n0 = kb * KBCODES + lx;        // subtile 0 code for this lane
    const int n1 = n0 + 16;                  // subtile 1 code
    const float esq0 = esq_g[n0];
    const float esq1 = esq_g[n1];
#pragma unroll
    for (int j = 0; j < 8; ++j) {
      float d0v = fmaf(-2.f, acc0[j], esq0);
      if (d0v < best[j]) { best[j] = d0v; bidx[j] = n0; }
      float d1v = fmaf(-2.f, acc1[j], esq1);
      if (d1v < best[j]) { best[j] = d1v; bidx[j] = n1; }
    }
  }

  // Cross-lane argmin within each 16-lane half (C layout: lanes 0-15 hold
  // rows 0-7, lanes 16-31 hold rows 8-15).  Tie-break on lowest index.
#pragma unroll
  for (int s = 8; s >= 1; s >>= 1) {
#pragma unroll
    for (int j = 0; j < 8; ++j) {
      float ov = __shfl_xor(best[j], s, 32);
      int   oi = __shfl_xor(bidx[j], s, 32);
      if (ov < best[j] || (ov == best[j] && oi < bidx[j])) { best[j] = ov; bidx[j] = oi; }
    }
  }
  if (lx == 0) {
#pragma unroll
    for (int j = 0; j < 8; ++j) {
      int m = m0 + wave * 16 + half * 8 + j;
      idx_g[m] = bidx[j];
      out_idx[m] = (float)bidx[j];
    }
  }
}

// ---------------------------------------------------------------------------
// Gather: quantized = E[idx] (full fp32), accumulate commitment-loss sum and
// per-code counts.  grid = NROWS blocks x 128 threads.
// ---------------------------------------------------------------------------
__global__ __launch_bounds__(128) void vq_gather(const float* __restrict__ x,
                                                 const float* __restrict__ e,
                                                 const int* __restrict__ idx_g,
                                                 float* __restrict__ q_out,
                                                 float* __restrict__ loss,
                                                 int* __restrict__ counts) {
  const int m = blockIdx.x;
  const int t = threadIdx.x;
  const int k = idx_g[m];
  float4 qv = ((const float4*)(e + (size_t)k * DIM))[t];
  float4 xv = ((const float4*)(x + (size_t)m * DIM))[t];
  ((float4*)(q_out + (size_t)m * DIM))[t] = qv;
  float dx = xv.x - qv.x, dy = xv.y - qv.y, dz = xv.z - qv.z, dw = xv.w - qv.w;
  float s = dx * dx + dy * dy + dz * dz + dw * dw;
#pragma unroll
  for (int o = 16; o >= 1; o >>= 1) s += __shfl_down(s, o, 32);
  __shared__ float r[4];
  if ((t & 31) == 0) r[t >> 5] = s;
  __syncthreads();
  if (t == 0) {
    atomicAdd(loss, r[0] + r[1] + r[2] + r[3]);
    atomicAdd(counts + k, 1);
  }
}

// ---------------------------------------------------------------------------
// Finalize: commitment loss scalar + perplexity.  1 block x 256 threads.
// ---------------------------------------------------------------------------
__global__ __launch_bounds__(256) void vq_finalize(const int* __restrict__ counts,
                                                   const float* __restrict__ loss,
                                                   float* __restrict__ out2) {
  const int t = threadIdx.x;
  float s = 0.f;
  for (int i = t; i < NUM_EMB; i += 256) {
    float p = (float)counts[i] * (1.f / (float)NROWS);
    s += p * logf(p + 1e-10f);
  }
#pragma unroll
  for (int o = 16; o >= 1; o >>= 1) s += __shfl_down(s, o, 32);
  __shared__ float r[8];
  if ((t & 31) == 0) r[t >> 5] = s;
  __syncthreads();
  if (t == 0) {
    float tot = 0.f;
#pragma unroll
    for (int i = 0; i < 8; ++i) tot += r[i];
    out2[1] = expf(-tot);                                    // perplexity
    out2[0] = 0.25f * (*loss) / (float)((size_t)NROWS * DIM); // commitment loss
  }
}

// ---------------------------------------------------------------------------
extern "C" void kernel_launch(void* const* d_in, const int* in_sizes, int n_in,
                              void* d_out, int out_size, void* d_ws, size_t ws_size,
                              hipStream_t stream) {
  (void)in_sizes; (void)n_in; (void)out_size; (void)ws_size;
  const float* x = (const float*)d_in[0];
  const float* e = (const float*)d_in[1];

  char* ws = (char*)d_ws;
  size_t off = 0;
  auto carve = [&](size_t bytes) -> void* {
    void* p = ws + off;
    off = (off + bytes + 255) & ~(size_t)255;
    return p;
  };
  float*  loss   = (float*)carve(sizeof(float));
  int*    counts = (int*)carve((size_t)NUM_EMB * sizeof(int));
  float*  esq    = (float*)carve((size_t)NUM_EMB * sizeof(float));
  int*    idx    = (int*)carve((size_t)NROWS * sizeof(int));
  bf16_t* ehi    = (bf16_t*)carve((size_t)NUM_EMB * DIM * sizeof(bf16_t));
  bf16_t* elo    = (bf16_t*)carve((size_t)NUM_EMB * DIM * sizeof(bf16_t));

  float* q_out   = (float*)d_out;                     // 32768*512 quantized
  float* scal    = q_out + (size_t)NROWS * DIM;       // [loss, perplexity]
  float* out_idx = scal + 2;                          // 32768 indices (as fp32)

  vq_prep<<<NUM_EMB, 128, 0, stream>>>(e, ehi, elo, esq, counts, loss);
  vq_argmin<<<NROWS / MTILE, 128, 0, stream>>>(x, ehi, elo, esq, idx, out_idx);
  vq_gather<<<NROWS, 128, 0, stream>>>(x, e, idx, q_out, loss, counts);
  vq_finalize<<<1, 256, 0, stream>>>(counts, loss, scal);
}
